// MultiHeadAttention_30794915512501
// MI455X (gfx1250) — matrix operand order repaired and checked by reading, not compiled, not run on hardware
//
#include <hip/hip_runtime.h>
#include <hip/hip_bf16.h>

#define BB 2
#define SS 2048
#define DD 1024
#define HH 16
#define HDD 64

typedef __attribute__((ext_vector_type(16))) __bf16 v16bf;
typedef __attribute__((ext_vector_type(8)))  float  v8f;

union V16BF { v16bf v; uint4 q[2]; };
union PK8   { uint4 q; __bf16 b[8]; };

__device__ __forceinline__ __bf16 f2bf(float f) {
    union { float f; unsigned u; } x; x.f = f;
    unsigned r = x.u + 0x7FFFu + ((x.u >> 16) & 1u);   // round-to-nearest-even
    unsigned short h = (unsigned short)(r >> 16);
    __bf16 o; __builtin_memcpy(&o, &h, 2);
    return o;
}

__device__ __forceinline__ v8f wmma_bf16(v16bf a, v16bf b, v8f c) {
    return __builtin_amdgcn_wmma_f32_16x16x32_bf16(
        /*neg_a=*/false, a, /*neg_b=*/false, b,
        /*c_mod=*/(short)0, c, /*reuse_a=*/false, /*reuse_b=*/false);
}

// A fragment (16x32 bf16): src row-major [M][K], rows m0..m0+15, cols k0..k0+31.
// Lane L (half=L>>4, m=L&15): elems 0..7 -> K = k0+half*8+0..7 ; elems 8..15 -> K = k0+16+half*8+0..7
__device__ __forceinline__ v16bf load_a(const __bf16* src, int ld, int m0, int k0) {
    int lane = threadIdx.x & 31;
    int half = lane >> 4, m = lane & 15;
    const __bf16* p = src + (size_t)(m0 + m) * ld + k0 + half * 8;
    V16BF r;
    r.q[0] = *(const uint4*)p;
    r.q[1] = *(const uint4*)(p + 16);
    return r.v;
}

// B fragment (32x16 bf16): src stored as [N][K] (each output column contiguous over K), ld = K stride.
// Lane L (half=L>>4, n=L&15): elems 0..7 -> K = k0+half*8+0..7 ; elems 8..15 -> K = k0+16+half*8+0..7
__device__ __forceinline__ v16bf load_bT(const __bf16* src, int ld, int n0, int k0) {
    int lane = threadIdx.x & 31;
    int half = lane >> 4, n = lane & 15;
    const __bf16* p = src + (size_t)(n0 + n) * ld + k0 + half * 8;
    V16BF r;
    r.q[0] = *(const uint4*)p;
    r.q[1] = *(const uint4*)(p + 16);
    return r.v;
}

__device__ __forceinline__ float redmax16(float x) {
    #pragma unroll
    for (int m = 1; m < 16; m <<= 1) x = fmaxf(x, __shfl_xor(x, m, 32));
    return x;
}
__device__ __forceinline__ float redsum16(float x) {
    #pragma unroll
    for (int m = 1; m < 16; m <<= 1) x += __shfl_xor(x, m, 32);
    return x;
}

// ---------------- Kernel 1: fp32 -> bf16 conversion (X, Wq, Wv) ----------------
__global__ void __launch_bounds__(256)
cvt_kernel(const float* __restrict__ emb, const float* __restrict__ wq,
           const float* __restrict__ wv, __bf16* __restrict__ xb,
           __bf16* __restrict__ wqb, __bf16* __restrict__ wvb) {
    size_t idx = (size_t)blockIdx.x * blockDim.x + threadIdx.x;
    const size_t nX = (size_t)BB * SS * DD;
    const size_t nW = (size_t)HH * HDD * DD;
    if (idx < nX) xb[idx] = f2bf(emb[idx]);
    if (idx < nW) { wqb[idx] = f2bf(wq[idx]); wvb[idx] = f2bf(wv[idx]); }
}

// ---------------- Kernel 2: Q/V projection GEMM ----------------
// Per wave: one (b,h,row-tile) -> Q tile [32 x 64] and V tile [32 x 64], K=D=1024.
// Two A fragments share every B fragment (weight reuse x2 vs 16-row tiles).
// Q stored row-major [B,H,S,HD]; V stored transposed [B,H,HD,S].
__global__ void __launch_bounds__(128)
proj_kernel(const __bf16* __restrict__ Xb, const __bf16* __restrict__ Wqb,
            const __bf16* __restrict__ Wvb, __bf16* __restrict__ Qb,
            __bf16* __restrict__ Vt) {
    const int NT = SS / 32;  // 64 row-tiles
    int w  = (blockIdx.x * blockDim.x + threadIdx.x) >> 5;
    int it = w % NT, h = (w / NT) % HH, b = w / (NT * HH);
    int s0 = it * 32;
    int lane = threadIdx.x & 31, half = lane >> 4, n = lane & 15;

    const __bf16* Wq = Wqb + (size_t)h * HDD * DD;
    const __bf16* Wv = Wvb + (size_t)h * HDD * DD;

    v8f zero = {};
    v8f aq[2][4] = {{zero, zero, zero, zero}, {zero, zero, zero, zero}};
    v8f av[2][4] = {{zero, zero, zero, zero}, {zero, zero, zero, zero}};

    for (int k = 0; k < DD; k += 32) {
        v16bf a0 = load_a(Xb, DD, b * SS + s0,      k);
        v16bf a1 = load_a(Xb, DD, b * SS + s0 + 16, k);
        #pragma unroll
        for (int c = 0; c < 4; c++) {
            v16bf bq = load_bT(Wq, DD, 16 * c, k);
            aq[0][c] = wmma_bf16(a0, bq, aq[0][c]);
            aq[1][c] = wmma_bf16(a1, bq, aq[1][c]);
            v16bf bv = load_bT(Wv, DD, 16 * c, k);
            av[0][c] = wmma_bf16(a0, bv, av[0][c]);
            av[1][c] = wmma_bf16(a1, bv, av[1][c]);
        }
    }

    size_t bh = (size_t)(b * HH + h);
    __bf16* qdst = Qb + (bh * SS + s0) * HDD;    // Q: row-major [S][HD]
    #pragma unroll
    for (int r = 0; r < 2; r++) {
        #pragma unroll
        for (int c = 0; c < 4; c++) {
            #pragma unroll
            for (int i = 0; i < 8; i++)
                qdst[(r * 16 + half * 8 + i) * HDD + 16 * c + n] = f2bf(aq[r][c][i]);
            PK8 pk;                               // V: transposed [HD][S], packed 16B stores
            #pragma unroll
            for (int i = 0; i < 8; i++) pk.b[i] = f2bf(av[r][c][i]);
            *(uint4*)(Vt + (bh * HDD + 16 * c + n) * SS + s0 + r * 16 + half * 8) = pk.q;
        }
    }
}

// ---------------- Kernel 3: causal flash attention (scores = Q·Qᵀ, out = softmax·V) ----------------
// 16 rows per wave, 64-column tiles: same WMMA count under causal masking as 32-wide,
// but half the cross-lane softmax/rescale overhead per column.
__global__ void __launch_bounds__(128)
attn_kernel(const __bf16* __restrict__ Qb, const __bf16* __restrict__ Vt,
            float* __restrict__ out) {
    __shared__ __align__(16) __bf16 plds[4][16 * 64];  // 2KB per wave P staging
    const int NT = SS / 16;
    int wslot = threadIdx.x >> 5;
    int w  = (blockIdx.x * blockDim.x + threadIdx.x) >> 5;
    int it = w % NT, h = (w / NT) % HH, b = w / (NT * HH);
    int s0 = it * 16;
    int lane = threadIdx.x & 31, half = lane >> 4, n = lane & 15;

    const __bf16* Qh = Qb + (size_t)(b * HH + h) * SS * HDD;
    const __bf16* Vh = Vt + (size_t)(b * HH + h) * HDD * SS;

    // Q row fragment is loop-invariant: rows s0..s0+15, K = e (0..63) -> two A frags
    v16bf qa0 = load_a(Qh, HDD, s0, 0);
    v16bf qa1 = load_a(Qh, HDD, s0, 32);

    v8f zero = {};
    v8f acc[4] = {zero, zero, zero, zero};
    float rm[8], rl[8], al[8];
    #pragma unroll
    for (int i = 0; i < 8; i++) { rm[i] = -3.0e38f; rl[i] = 0.0f; }

    int njt = (s0 >> 6) + 1;  // causal: 64-wide column tiles up to the diagonal
    for (int j = 0; j < njt; j++) {
        int t0 = j * 64;

        // prefetch next tile's Q columns (rows t0+64..t0+127) and V rows into cache
        if (j + 1 < njt) {
            int tn = t0 + 64;
            __builtin_prefetch(Qh + (size_t)(tn + lane) * HDD, 0, 3);
            __builtin_prefetch(Qh + (size_t)(tn + 32 + lane) * HDD, 0, 3);
            __builtin_prefetch(Vh + (size_t)lane * SS + tn, 0, 3);
            __builtin_prefetch(Vh + (size_t)(lane + 32) * SS + tn, 0, 3);
        }

        // ---- scores tile 16x64 = four 16x16 C tiles, K = HD = 64 ----
        v8f sc[4] = {zero, zero, zero, zero};
        #pragma unroll
        for (int c = 0; c < 4; c++) {
            sc[c] = wmma_bf16(qa0, load_bT(Qh, HDD, t0 + 16 * c, 0),  sc[c]);
            sc[c] = wmma_bf16(qa1, load_bT(Qh, HDD, t0 + 16 * c, 32), sc[c]);
        }

        // ---- scale + causal mask ----
        float p[4][8];
        #pragma unroll
        for (int c = 0; c < 4; c++)
            #pragma unroll
            for (int i = 0; i < 8; i++) {
                int srow = s0 + half * 8 + i;
                float x = sc[c][i] * 0.125f;
                if (t0 + 16 * c + n > srow) x = -3.0e38f;
                p[c][i] = x;
            }

        // ---- online softmax update (row stats across the 16 lanes of each half) ----
        #pragma unroll
        for (int i = 0; i < 8; i++) {
            float mx = fmaxf(fmaxf(p[0][i], p[1][i]), fmaxf(p[2][i], p[3][i]));
            mx = redmax16(mx);
            float mn = fmaxf(rm[i], mx);
            float a  = __expf(rm[i] - mn);
            float s  = 0.0f;
            #pragma unroll
            for (int c = 0; c < 4; c++) { p[c][i] = __expf(p[c][i] - mn); s += p[c][i]; }
            rl[i] = rl[i] * a + redsum16(s);
            rm[i] = mn; al[i] = a;
        }

        // ---- P (C layout) -> LDS -> two A fragments ----
        __bf16* pl = &plds[wslot][0];
        #pragma unroll
        for (int c = 0; c < 4; c++)
            #pragma unroll
            for (int i = 0; i < 8; i++)
                pl[(half * 8 + i) * 64 + 16 * c + n] = f2bf(p[c][i]);

        V16BF pf0, pf1;
        const __bf16* pa = pl + (lane & 15) * 64 + half * 8;
        pf0.q[0] = *(const uint4*)pa;
        pf0.q[1] = *(const uint4*)(pa + 16);
        pf1.q[0] = *(const uint4*)(pa + 32);
        pf1.q[1] = *(const uint4*)(pa + 48);

        // ---- acc = acc*alpha + P @ V_tile  (K = 64 over t) ----
        #pragma unroll
        for (int c = 0; c < 4; c++) {
            #pragma unroll
            for (int i = 0; i < 8; i++) acc[c][i] *= al[i];
            acc[c] = wmma_bf16(pf0.v, load_bT(Vh, SS, 16 * c, t0),      acc[c]);
            acc[c] = wmma_bf16(pf1.v, load_bT(Vh, SS, 16 * c, t0 + 32), acc[c]);
        }
    }

    // ---- epilogue: out[b, s, h*HD + e] = acc / l ----
    #pragma unroll
    for (int i = 0; i < 8; i++) {
        float inv = 1.0f / rl[i];
        int srow = s0 + half * 8 + i;
        float* orow = out + ((size_t)b * SS + srow) * DD + h * HDD;
        #pragma unroll
        for (int c = 0; c < 4; c++) orow[16 * c + n] = acc[c][i] * inv;
    }
}

extern "C" void kernel_launch(void* const* d_in, const int* in_sizes, int n_in,
                              void* d_out, int out_size, void* d_ws, size_t ws_size,
                              hipStream_t stream) {
    (void)in_sizes; (void)n_in; (void)out_size; (void)ws_size;
    const float* emb = (const float*)d_in[0];
    const float* wq  = (const float*)d_in[1];
    // d_in[2] = Wk: dead code in the reference output — skipped.
    const float* wv  = (const float*)d_in[3];
    float* out = (float*)d_out;

    char* ws = (char*)d_ws;
    const size_t nX = (size_t)BB * SS * DD;       // 4M elems
    const size_t nW = (size_t)HH * HDD * DD;      // 1M elems
    const size_t nQ = (size_t)BB * HH * SS * HDD; // 4M elems
    __bf16* Xb  = (__bf16*)(ws);
    __bf16* Wqb = (__bf16*)(ws + nX * 2);
    __bf16* Wvb = (__bf16*)(ws + nX * 2 + nW * 2);
    __bf16* Qb  = (__bf16*)(ws + nX * 2 + nW * 4);
    __bf16* Vt  = (__bf16*)(ws + nX * 2 + nW * 4 + nQ * 2);  // total 28 MB

    int cthreads = 256;
    int cblocks  = (int)((nX + cthreads - 1) / cthreads);
    cvt_kernel<<<cblocks, cthreads, 0, stream>>>(emb, wq, wv, Xb, Wqb, Wvb);

    int pwaves = BB * HH * (SS / 32);   // 2048 wave-units (32 rows each)
    proj_kernel<<<pwaves / 4, 128, 0, stream>>>(Xb, Wqb, Wvb, Qb, Vt);

    int awaves = BB * HH * (SS / 16);   // 4096 wave-units (16 rows each)
    attn_kernel<<<awaves / 4, 128, 0, stream>>>(Qb, Vt, out);
}
